// ROIhead_87299505258966
// MI455X (gfx1250) — matrix operand order repaired and checked by reading, not compiled, not run on hardware
//
#include <hip/hip_runtime.h>
#include <hip/hip_bf16.h>
#include <math.h>

// ---------------- problem constants ----------------
#define N_PROP      1000
#define M_PAD       1024          // proposals padded to multiple of 256
#define NUM_CLASSES 21
#define POOLSZ      7
#define FC_DIM      4096
#define C_FEAT      512
#define K_FC6       (C_FEAT * POOLSZ * POOLSZ)   // 25088
#define HFEAT       50
#define WFEAT       50
#define IMG_WH      800.0f
#define SCORE_THR   0.05f
#define NMS_THR     0.3f
#define TOP_N       100
#define DW_CLAMP    4.135166556742356f           // log(1000/16)

#ifndef __has_builtin
#define __has_builtin(x) 0
#endif
#if __has_builtin(__builtin_amdgcn_tensor_load_to_lds) && __has_builtin(__builtin_amdgcn_s_wait_tensorcnt)
#define USE_TDM 1
#else
#define USE_TDM 0
#endif

// ---------------- vector types ----------------
typedef __attribute__((ext_vector_type(16))) _Float16     v16h;
typedef __attribute__((ext_vector_type(8)))  _Float16     v8h;
typedef __attribute__((ext_vector_type(4)))  _Float16     v4h;
typedef __attribute__((ext_vector_type(8)))  float        v8f;
typedef __attribute__((ext_vector_type(4)))  float        v4f;
typedef __attribute__((ext_vector_type(4)))  unsigned int v4u;
typedef __attribute__((ext_vector_type(8)))  int          v8i;
typedef __attribute__((ext_vector_type(4)))  int          v4i;

__device__ __forceinline__ v16h cat8(v8h lo, v8h hi) {
  return __builtin_shufflevector(lo, hi, 0,1,2,3,4,5,6,7,8,9,10,11,12,13,14,15);
}
__device__ __forceinline__ v8h cat4(v4h lo, v4h hi) {
  return __builtin_shufflevector(lo, hi, 0,1,2,3,4,5,6,7);
}

#if USE_TDM
// ---------------------------------------------------------------------
// TDM: DMA a tile of 256 rows x 64 bytes (32 f16) from global into LDS.
//  data_size = 3 (8-byte units); tile_dim0 = 8 units (64B); tile_dim1 = 256
//  tensor_dim0_stride = row stride in 8B units. Descriptor D# per ISA 8.3/8.4.
// ---------------------------------------------------------------------
__device__ __forceinline__ void tdm_load_tile(unsigned lds_addr, const void* gptr,
                                              unsigned row_stride_units) {
  unsigned long long ga = (unsigned long long)gptr;
  v4u g0;
  g0.x = 1u;                                          // count=1, user mode
  g0.y = lds_addr;                                    // lds_addr [63:32]
  g0.z = (unsigned)(ga & 0xFFFFFFFFu);                // global_addr lo
  g0.w = (unsigned)((ga >> 32) & 0x01FFFFFFu)         // global_addr hi
       | (2u << 30);                                  // type = 2 (image)
  const unsigned td0 = 0x003FFFFFu;                   // large tensor dims (no OOB)
  const unsigned td1 = 0x003FFFFFu;
  v8i g1;
  g1[0] = 0x00030000;                                 // wg_mask=0, data_size=3 (8B)
  g1[1] = (int)((td0 & 0xFFFFu) << 16);               // tensor_dim0 [79:48] lo
  g1[2] = (int)((td0 >> 16) | ((td1 & 0xFFFFu) << 16)); // td0 hi | tensor_dim1 lo
  g1[3] = (int)((td1 >> 16) | (8u << 16));            // td1 hi | tile_dim0 = 8
  g1[4] = 256;                                        // tile_dim1 = 256, tile_dim2 = 0
  g1[5] = (int)row_stride_units;                      // tensor_dim0_stride lo32
  g1[6] = 0;                                          // stride hi | dim1_stride lo
  g1[7] = 0;
  v4i z4 = {0, 0, 0, 0};
#if defined(__clang_major__) && __clang_major__ >= 23
  v8i z8 = {0, 0, 0, 0, 0, 0, 0, 0};
  __builtin_amdgcn_tensor_load_to_lds(g0, g1, z4, z4, z8, 0);
#else
  __builtin_amdgcn_tensor_load_to_lds(g0, g1, z4, z4, 0);
#endif
}
#endif // USE_TDM

// =====================================================================
// Kernel 1: ROI max pooling -> A[M_PAD, 25088] in f16 (pad rows zeroed)
// =====================================================================
__global__ __launch_bounds__(256)
void roipool_kernel(const float* __restrict__ fm, const float* __restrict__ props,
                    _Float16* __restrict__ A) {
  int p = blockIdx.x;
  if (p >= N_PROP) {
    for (int idx = threadIdx.x; idx < K_FC6; idx += 256)
      A[(size_t)p * K_FC6 + idx] = (_Float16)0.0f;
    return;
  }
  float x1 = props[p * 4 + 0], y1 = props[p * 4 + 1];
  float x2 = props[p * 4 + 2], y2 = props[p * 4 + 3];
  float rs_w = rintf(x1 * 0.0625f), rs_h = rintf(y1 * 0.0625f);
  float re_w = rintf(x2 * 0.0625f), re_h = rintf(y2 * 0.0625f);
  float roi_w = fmaxf(re_w - rs_w, 1.0f);
  float roi_h = fmaxf(re_h - rs_h, 1.0f);

  for (int idx = threadIdx.x; idx < K_FC6; idx += 256) {
    int c   = idx / 49;
    int bin = idx - c * 49;
    int bh  = bin / 7, bw = bin - bh * 7;
    int hs0 = (int)fminf(fmaxf(floorf((float)bh       * roi_h * (1.0f/7.0f)) + rs_h, 0.0f), 50.0f);
    int he0 = (int)fminf(fmaxf(ceilf ((float)(bh + 1) * roi_h * (1.0f/7.0f)) + rs_h, 0.0f), 50.0f);
    int ws0 = (int)fminf(fmaxf(floorf((float)bw       * roi_w * (1.0f/7.0f)) + rs_w, 0.0f), 50.0f);
    int we0 = (int)fminf(fmaxf(ceilf ((float)(bw + 1) * roi_w * (1.0f/7.0f)) + rs_w, 0.0f), 50.0f);
    float mx = -INFINITY;
    const float* fc = fm + (size_t)c * (HFEAT * WFEAT);
    for (int h = hs0; h < he0; ++h)
      for (int w = ws0; w < we0; ++w)
        mx = fmaxf(mx, fc[h * WFEAT + w]);
    if (isinf(mx)) mx = 0.0f;
    A[(size_t)p * K_FC6 + idx] = (_Float16)mx;
  }
}

// =====================================================================
// Kernel 1b: one-shot fp32 -> f16 weight conversion (bandwidth bound)
// =====================================================================
__global__ __launch_bounds__(256)
void cvt_w_kernel(const float* __restrict__ src, _Float16* __restrict__ dst, long long n) {
  long long i = ((long long)blockIdx.x * 256 + threadIdx.x) * 8;
  if (i + 8 <= n) {
    v4f f0 = *(const v4f*)(src + i);
    v4f f1 = *(const v4f*)(src + i + 4);
    *(v8h*)(dst + i) = cat4(__builtin_convertvector(f0, v4h),
                            __builtin_convertvector(f1, v4h));
  }
}

// =====================================================================
// Kernel 2/3: WMMA GEMM:  C[M,N] = relu(A[M,K] * Wh[N,K]^T + bias)
//   A, Wh: f16 row-major. Block: 8 waves, tile 256(M) x 128(N).
//   Wave: 64x64 = 4x4 v_wmma_f32_16x16x32_f16 tiles.
//   A tile double-buffered in LDS; staged by the Tensor Data Mover
//   (tensor_load_to_lds, TENSORcnt) overlapping the WMMA compute.
// =====================================================================
__global__ __launch_bounds__(256)
void gemm_f16_wmma(const _Float16* __restrict__ A, const _Float16* __restrict__ Wh,
                   const float* __restrict__ bias, _Float16* __restrict__ C,
                   int K, int N, int relu) {
  __shared__ _Float16 As[2][256 * 32];     // 2 x 16 KB double buffer

  const int tid    = threadIdx.x;
  const int lane   = tid & 31;
  const int wave   = tid >> 5;
  const int wm     = wave & 3;             // 4 M-groups
  const int wn     = wave >> 2;            // 2 N-groups
  const int laneN  = lane & 15;
  const int laneHi = lane >> 4;            // 0 | 1
  const int m_block = blockIdx.y * 256;
  const int n_wave  = blockIdx.x * 128 + wn * 64;

  v8f acc[4][4] = {};

  // B fragment: lane holds column n, 16 consecutive K at k0 + laneHi*16
  const _Float16* wbase = Wh + (size_t)(n_wave + laneN) * K + laneHi * 16;
  // A fragment: lane<16 row=laneN K {0..15};
  // lane>=16 K {16..31}
  const int a_row  = wm * 64 + laneN;
  const int a_koff = laneHi * 16;

  // ---- staging of one 256x32-f16 tile into LDS buffer `buf` ----
  auto stage = [&](int buf, int k0) {
#if USE_TDM
    if (wave == 0) {
      unsigned lbase = (unsigned)(size_t)(&As[buf][0]);
      tdm_load_tile(lbase, A + (size_t)m_block * K + k0, (unsigned)(K >> 2));
    }
#else
    #pragma unroll
    for (int ch = 0; ch < 4; ++ch) {
      int id = tid + ch * 256;
      int r  = id >> 2;
      int co = (id & 3) * 8;
      *(v8h*)(&As[buf][r * 32 + co]) =
          *(const v8h*)(A + (size_t)(m_block + r) * K + k0 + co);
    }
#endif
  };
  auto stage_wait = [&]() {
#if USE_TDM
    if (wave == 0) __builtin_amdgcn_s_wait_tensorcnt(0);
#endif
  };

  // prologue: fill buffer 0
  stage(0, 0);
  stage_wait();
  __syncthreads();

  int cur = 0;
  for (int k0 = 0; k0 < K; k0 += 32) {
    // kick DMA for the next tile while we compute on this one
    if (k0 + 32 < K) stage(cur ^ 1, k0 + 32);

    // ---- B fragments: pure f16 b128 loads, no conversion ----
    v16h bf[4];
    #pragma unroll
    for (int ni = 0; ni < 4; ++ni) {
      const _Float16* wp = wbase + (size_t)ni * 16 * K + k0;
      __builtin_prefetch(wp + 32, 0, 1);          // global_prefetch next k-slab
      v8h lo = *(const v8h*)(wp);
      v8h hi = *(const v8h*)(wp + 8);
      bf[ni] = cat8(lo, hi);
    }

    // ---- A fragments from LDS ----
    v16h af[4];
    #pragma unroll
    for (int mi = 0; mi < 4; ++mi) {
      const _Float16* ap = &As[cur][(a_row + mi * 16) * 32 + a_koff];
      af[mi] = cat8(*(const v8h*)ap, *(const v8h*)(ap + 8));
    }

    // ---- 16 WMMAs ----
    #pragma unroll
    for (int mi = 0; mi < 4; ++mi)
      #pragma unroll
      for (int ni = 0; ni < 4; ++ni)
        acc[mi][ni] = __builtin_amdgcn_wmma_f32_16x16x32_f16(
            false, af[mi], false, bf[ni], (short)0, acc[mi][ni], false, false);

    stage_wait();      // TENSORcnt drain lands after compute
    __syncthreads();
    cur ^= 1;
  }

  // ---- epilogue: bias + relu + f16 store (C layout: M = r + 8*laneHi) ----
  #pragma unroll
  for (int ni = 0; ni < 4; ++ni) {
    int n = n_wave + ni * 16 + laneN;
    float b = bias[n];
    #pragma unroll
    for (int mi = 0; mi < 4; ++mi) {
      #pragma unroll
      for (int r = 0; r < 8; ++r) {
        int m = m_block + wm * 64 + mi * 16 + laneHi * 8 + r;
        float v = acc[mi][ni][r] + b;
        if (relu) v = fmaxf(v, 0.0f);
        C[(size_t)m * N + n] = (_Float16)v;
      }
    }
  }
}

// =====================================================================
// Kernel 4: cls + reg heads. O[p][j], j<21 -> cls, j in 21..104 -> reg
// =====================================================================
__global__ __launch_bounds__(256)
void heads_kernel(const _Float16* __restrict__ X, const float* __restrict__ w_cls,
                  const float* __restrict__ b_cls, const float* __restrict__ w_reg,
                  const float* __restrict__ b_reg, float* __restrict__ O) {
  int idx = blockIdx.x * blockDim.x + threadIdx.x;
  if (idx >= N_PROP * 105) return;
  int m = idx / 105, j = idx - m * 105;
  const float* wrow;
  float b;
  if (j < 21) { wrow = w_cls + (size_t)j * FC_DIM;        b = b_cls[j]; }
  else        { wrow = w_reg + (size_t)(j - 21) * FC_DIM; b = b_reg[j - 21]; }
  const _Float16* x = X + (size_t)m * FC_DIM;
  float s = b;
  for (int k = 0; k < FC_DIM; k += 4) {
    s += (float)x[k]     * wrow[k]     + (float)x[k + 1] * wrow[k + 1]
       + (float)x[k + 2] * wrow[k + 2] + (float)x[k + 3] * wrow[k + 3];
  }
  O[idx] = s;
}

// =====================================================================
// Kernel 5: softmax + box decode (per proposal, classes 1..20)
// =====================================================================
__global__ void decode_kernel(const float* __restrict__ O, const float* __restrict__ props,
                              float* __restrict__ BXS, float* __restrict__ SCS) {
  int p = blockIdx.x;
  int t = threadIdx.x;
  __shared__ float probs[NUM_CLASSES];
  const float* cls = O + (size_t)p * 105;
  if (t == 0) {
    float mx = cls[0];
    for (int c = 1; c < NUM_CLASSES; ++c) mx = fmaxf(mx, cls[c]);
    float sum = 0.0f;
    for (int c = 0; c < NUM_CLASSES; ++c) { float e = expf(cls[c] - mx); probs[c] = e; sum += e; }
    float inv = 1.0f / sum;
    for (int c = 0; c < NUM_CLASSES; ++c) probs[c] *= inv;
  }
  __syncthreads();
  if (t < 20) {
    int c = t + 1;
    const float* reg = O + (size_t)p * 105 + 21 + c * 4;
    float dx = reg[0] * 0.1f, dy = reg[1] * 0.1f;
    float dw = fminf(reg[2] * 0.2f, DW_CLAMP);
    float dh = fminf(reg[3] * 0.2f, DW_CLAMP);
    float px1 = props[p * 4], py1 = props[p * 4 + 1];
    float px2 = props[p * 4 + 2], py2 = props[p * 4 + 3];
    float pw = px2 - px1, ph = py2 - py1;
    float pcx = px1 + 0.5f * pw, pcy = py1 + 0.5f * ph;
    float cx = dx * pw + pcx, cy = dy * ph + pcy;
    float bw = expf(dw) * pw, bh = expf(dh) * ph;
    float bx1 = fminf(fmaxf(cx - 0.5f * bw, 0.0f), IMG_WH);
    float by1 = fminf(fmaxf(cy - 0.5f * bh, 0.0f), IMG_WH);
    float bx2 = fminf(fmaxf(cx + 0.5f * bw, 0.0f), IMG_WH);
    float by2 = fminf(fmaxf(cy + 0.5f * bh, 0.0f), IMG_WH);
    float* bp = BXS + ((size_t)p * 20 + t) * 4;
    bp[0] = bx1; bp[1] = by1; bp[2] = bx2; bp[3] = by2;
    SCS[(size_t)p * 20 + t] = probs[c];
  }
}

// =====================================================================
// Kernel 6: per-class greedy NMS (20 blocks, LDS-resident)
// =====================================================================
__global__ __launch_bounds__(256)
void nms_kernel(const float* __restrict__ BXS, const float* __restrict__ SCS,
                float* __restrict__ MSK) {
  int c   = blockIdx.x;
  int tid = threadIdx.x;
  __shared__ float bx[N_PROP][4];
  __shared__ float sc[N_PROP];
  __shared__ int   st[N_PROP];
  __shared__ float rmax[256];
  __shared__ int   rarg[256];
  __shared__ int   s_done;
  __shared__ float s_box[4];

  for (int i = tid; i < N_PROP; i += 256) {
    const float* bp = BXS + ((size_t)i * 20 + c) * 4;
    float b0 = bp[0], b1 = bp[1], b2 = bp[2], b3 = bp[3];
    bx[i][0] = b0; bx[i][1] = b1; bx[i][2] = b2; bx[i][3] = b3;
    float s = SCS[(size_t)i * 20 + c];
    sc[i] = s;
    st[i] = (s > SCORE_THR && (b2 - b0) >= 1.0f && (b3 - b1) >= 1.0f) ? 0 : 1;
  }
  __syncthreads();

  for (int it = 0; it < N_PROP; ++it) {
    float best = -1e30f; int barg = -1;
    for (int i = tid; i < N_PROP; i += 256)
      if (st[i] == 0 && sc[i] > best) { best = sc[i]; barg = i; }
    rmax[tid] = best; rarg[tid] = barg;
    __syncthreads();
    for (int off = 128; off > 0; off >>= 1) {
      if (tid < off && rmax[tid + off] > rmax[tid]) {
        rmax[tid] = rmax[tid + off]; rarg[tid] = rarg[tid + off];
      }
      __syncthreads();
    }
    if (tid == 0) {
      int b = rarg[0];
      if (b < 0) { s_done = 1; }
      else {
        s_done = 0;
        st[b] = 2;
        s_box[0] = bx[b][0]; s_box[1] = bx[b][1];
        s_box[2] = bx[b][2]; s_box[3] = bx[b][3];
      }
    }
    __syncthreads();
    if (s_done) break;
    float ax1 = s_box[0], ay1 = s_box[1], ax2 = s_box[2], ay2 = s_box[3];
    float aarea = (ax2 - ax1) * (ay2 - ay1);
    for (int i = tid; i < N_PROP; i += 256) {
      if (st[i] != 0) continue;
      float ix1 = fmaxf(ax1, bx[i][0]);
      float iy1 = fmaxf(ay1, bx[i][1]);
      float ix2 = fminf(ax2, bx[i][2]);
      float iy2 = fminf(ay2, bx[i][3]);
      float inter = fmaxf(ix2 - ix1, 0.0f) * fmaxf(iy2 - iy1, 0.0f);
      float barea = (bx[i][2] - bx[i][0]) * (bx[i][3] - bx[i][1]);
      float iou = inter / (aarea + barea - inter);
      if (iou > NMS_THR) st[i] = 1;
    }
    __syncthreads();
  }

  for (int i = tid; i < N_PROP; i += 256)
    MSK[(size_t)i * 20 + c] = (st[i] == 2) ? sc[i] : -1.0f;
}

// =====================================================================
// Kernel 7: iterative top-100. out: [100*4 boxes][100 labels][100 scores]
// =====================================================================
__global__ __launch_bounds__(256)
void topk_kernel(float* __restrict__ MSK, const float* __restrict__ BXS,
                 float* __restrict__ out) {
  int tid = threadIdx.x;
  __shared__ float rmax[256];
  __shared__ int   rarg[256];
  const int TOT = N_PROP * 20;
  for (int sel = 0; sel < TOP_N; ++sel) {
    float best = -1e30f; int barg = 0;
    for (int i = tid; i < TOT; i += 256)
      if (MSK[i] > best) { best = MSK[i]; barg = i; }
    rmax[tid] = best; rarg[tid] = barg;
    __syncthreads();
    for (int off = 128; off > 0; off >>= 1) {
      if (tid < off && rmax[tid + off] > rmax[tid]) {
        rmax[tid] = rmax[tid + off]; rarg[tid] = rarg[tid + off];
      }
      __syncthreads();
    }
    if (tid == 0) {
      int b = rarg[0];
      int cls = b - (b / 20) * 20;
      out[sel * 4 + 0] = BXS[(size_t)b * 4 + 0];
      out[sel * 4 + 1] = BXS[(size_t)b * 4 + 1];
      out[sel * 4 + 2] = BXS[(size_t)b * 4 + 2];
      out[sel * 4 + 3] = BXS[(size_t)b * 4 + 3];
      out[4 * TOP_N + sel] = (float)(cls + 1);
      out[5 * TOP_N + sel] = rmax[0];
      MSK[b] = -1e30f;           // nms_kernel fully rewrites MSK each call
    }
    __syncthreads();
  }
}

// =====================================================================
// Host launcher
// =====================================================================
extern "C" void kernel_launch(void* const* d_in, const int* in_sizes, int n_in,
                              void* d_out, int out_size, void* d_ws, size_t ws_size,
                              hipStream_t stream) {
  const float* feature_map = (const float*)d_in[0];
  const float* proposals   = (const float*)d_in[1];
  const float* w_fc6       = (const float*)d_in[2];
  const float* b_fc6       = (const float*)d_in[3];
  const float* w_fc7       = (const float*)d_in[4];
  const float* b_fc7       = (const float*)d_in[5];
  const float* w_cls       = (const float*)d_in[6];
  const float* b_cls       = (const float*)d_in[7];
  const float* w_reg       = (const float*)d_in[8];
  const float* b_reg       = (const float*)d_in[9];
  float* out = (float*)d_out;

  char* ws = (char*)d_ws;
  size_t off = 0;
  auto carve = [&](size_t bytes) -> char* {
    char* p = ws + off;
    off += bytes;
    off = (off + 255) & ~(size_t)255;
    return p;
  };
  _Float16* A6   = (_Float16*)carve((size_t)M_PAD * K_FC6 * sizeof(_Float16));
  _Float16* W6h  = (_Float16*)carve((size_t)FC_DIM * K_FC6 * sizeof(_Float16));
  _Float16* W7h  = (_Float16*)carve((size_t)FC_DIM * FC_DIM * sizeof(_Float16));
  _Float16* X1   = (_Float16*)carve((size_t)M_PAD * FC_DIM * sizeof(_Float16));
  _Float16* X2   = (_Float16*)carve((size_t)M_PAD * FC_DIM * sizeof(_Float16));
  float*    O105 = (float*)carve((size_t)N_PROP * 105 * sizeof(float));
  float*    BXS  = (float*)carve((size_t)N_PROP * 20 * 4 * sizeof(float));
  float*    SCS  = (float*)carve((size_t)N_PROP * 20 * sizeof(float));
  float*    MSK  = (float*)carve((size_t)N_PROP * 20 * sizeof(float));

  // weight pre-conversion (one pass, bandwidth bound)
  {
    long long n6 = (long long)FC_DIM * K_FC6;
    long long n7 = (long long)FC_DIM * FC_DIM;
    cvt_w_kernel<<<(unsigned)((n6 / 8 + 255) / 256), 256, 0, stream>>>(w_fc6, W6h, n6);
    cvt_w_kernel<<<(unsigned)((n7 / 8 + 255) / 256), 256, 0, stream>>>(w_fc7, W7h, n7);
  }

  roipool_kernel<<<M_PAD, 256, 0, stream>>>(feature_map, proposals, A6);

  dim3 g6(FC_DIM / 128, M_PAD / 256);
  gemm_f16_wmma<<<g6, 256, 0, stream>>>(A6, W6h, b_fc6, X1, K_FC6, FC_DIM, 1);
  gemm_f16_wmma<<<g6, 256, 0, stream>>>(X1, W7h, b_fc7, X2, FC_DIM, FC_DIM, 1);

  heads_kernel<<<(N_PROP * 105 + 255) / 256, 256, 0, stream>>>(X2, w_cls, b_cls, w_reg, b_reg, O105);
  decode_kernel<<<N_PROP, 32, 0, stream>>>(O105, proposals, BXS, SCS);
  nms_kernel<<<20, 256, 0, stream>>>(BXS, SCS, MSK);
  topk_kernel<<<1, 256, 0, stream>>>(MSK, BXS, out);
}
